// ChangeGuideModule_60344290509024
// MI455X (gfx1250) — compile-verified, matrix operand order checked by reading, not run on hardware
//
#include <hip/hip_runtime.h>

// ---------------------------------------------------------------------------
// Types for CDNA5 WMMA (wave32).
// ---------------------------------------------------------------------------
typedef __attribute__((ext_vector_type(16))) __bf16         v16bf;
typedef __attribute__((ext_vector_type(8)))  float          v8f;
typedef __attribute__((ext_vector_type(8)))  unsigned short u16x8;

struct BFrag {            // 16 bf16 elements = 8 VGPRs, as two 16-byte halves
    u16x8 lo;             // elements e=0..7   (K = 8*half + e)
    u16x8 hi;             // elements e=8..15  (K = 16 + 8*half + (e-8))
};

static __device__ inline v8f wmma_bf16(const BFrag& a, const BFrag& b, v8f c) {
    return __builtin_amdgcn_wmma_f32_16x16x32_bf16(
        false, __builtin_bit_cast(v16bf, a),
        false, __builtin_bit_cast(v16bf, b),
        (short)0, c, false, false);
}

static __device__ __host__ inline unsigned short f2bf(float f) {
#if defined(__HIP_DEVICE_COMPILE__)
    return __builtin_bit_cast(unsigned short, static_cast<__bf16>(f));
#else
    union { float f; unsigned u; } cvt; cvt.f = f;
    unsigned r = cvt.u + 0x7FFFu + ((cvt.u >> 16) & 1u);
    return (unsigned short)(r >> 16);
#endif
}

static __device__ inline float redmax16(float v) {
    #pragma unroll
    for (int off = 1; off < 16; off <<= 1) v = fmaxf(v, __shfl_xor(v, off, 32));
    return v;
}
static __device__ inline float redsum16(float v) {
    #pragma unroll
    for (int off = 1; off < 16; off <<= 1) v += __shfl_xor(v, off, 32);
    return v;
}

#define B_DIM   4
#define C_DIM   256
#define C8_DIM  32
#define NTOK    4096      // 64*64
#define MTILE   32        // key tile
#define QBLK    64        // query block per workgroup

// ---------------------------------------------------------------------------
// Kernel 1: bilinear(align_corners=True) 32x32 -> 64x64, gate = 1 + sigmoid.
// ---------------------------------------------------------------------------
__global__ void gate_kernel(const float* __restrict__ gm, float* __restrict__ gate) {
    int idx = blockIdx.x * blockDim.x + threadIdx.x;
    if (idx >= B_DIM * NTOK) return;
    int b = idx >> 12;
    int n = idx & (NTOK - 1);
    int oy = n >> 6, ox = n & 63;
    float fy = oy * (31.0f / 63.0f);
    float fx = ox * (31.0f / 63.0f);
    int y0 = (int)fy, x0 = (int)fx;
    int y1 = y0 + 1 > 31 ? 31 : y0 + 1;
    int x1 = x0 + 1 > 31 ? 31 : x0 + 1;
    float wy = fy - (float)y0, wx = fx - (float)x0;
    const float* g = gm + b * 1024;
    float v00 = g[y0 * 32 + x0], v01 = g[y0 * 32 + x1];
    float v10 = g[y1 * 32 + x0], v11 = g[y1 * 32 + x1];
    float top = v00 * (1.0f - wx) + v01 * wx;
    float bot = v10 * (1.0f - wx) + v11 * wx;
    float z   = top * (1.0f - wy) + bot * wy;
    gate[idx] = 1.0f + 1.0f / (1.0f + __expf(-z));
}

// ---------------------------------------------------------------------------
// Kernel 2: fused QKV projection via WMMA.
//   rows 0..31   -> Q -> qbuf[B][N][32]  bf16 (token-major)
//   rows 32..63  -> K -> kbuf[B][N][32]  bf16
//   rows 64..319 -> V -> vbuf[B][256][N] bf16 (channel-major)
// ---------------------------------------------------------------------------
__global__ __launch_bounds__(32) void qkv_kernel(
    const float* __restrict__ x,
    const float* __restrict__ Wq, const float* __restrict__ bq,
    const float* __restrict__ Wk, const float* __restrict__ bk,
    const float* __restrict__ Wv, const float* __restrict__ bv,
    const float* __restrict__ gate,
    unsigned short* __restrict__ qbuf,
    unsigned short* __restrict__ kbuf,
    unsigned short* __restrict__ vbuf)
{
    const int b    = blockIdx.z;
    const int rt   = blockIdx.y;            // 0..19 row tiles
    const int n0   = blockIdx.x * 16;       // token tile
    const int lane = threadIdx.x & 31;
    const int half = lane >> 4;
    const int l15  = lane & 15;
    const int row_base = rt * 16;

    const float* W; const float* bias; int wrow;
    if (row_base < 32)       { W = Wq; bias = bq; wrow = row_base;      }
    else if (row_base < 64)  { W = Wk; bias = bk; wrow = row_base - 32; }
    else                     { W = Wv; bias = bv; wrow = row_base - 64; }

    const int n = n0 + l15;
    v8f acc = {};
    for (int k0 = 0; k0 < C_DIM; k0 += 32) {
        BFrag af, bf;
        const float* wp = W + (size_t)(wrow + l15) * C_DIM + k0;
        #pragma unroll
        for (int e = 0; e < 8; ++e) {
            af.lo[e] = f2bf(wp[8 * half + e]);
            af.hi[e] = f2bf(wp[16 + 8 * half + e]);
        }
        const float* xp = x + ((size_t)b * C_DIM + k0) * NTOK + n;
        #pragma unroll
        for (int e = 0; e < 8; ++e) {
            bf.lo[e] = f2bf(xp[(size_t)(8 * half + e) * NTOK]);
            bf.hi[e] = f2bf(xp[(size_t)(16 + 8 * half + e) * NTOK]);
        }
        acc = wmma_bf16(af, bf, acc);
    }

    const float gt = gate[b * NTOK + n];
    if (row_base < 64) {
        unsigned short* dst = (row_base < 32 ? qbuf : kbuf)
                            + ((size_t)b * NTOK + n) * C8_DIM + wrow + 8 * half;
        u16x8 pk;
        #pragma unroll
        for (int r = 0; r < 8; ++r)
            pk[r] = f2bf((acc[r] + bias[wrow + r + 8 * half]) * gt);
        *(u16x8*)dst = pk;
    } else {
        #pragma unroll
        for (int r = 0; r < 8; ++r) {
            int c = wrow + r + 8 * half;
            vbuf[((size_t)b * C_DIM + c) * NTOK + n] =
                f2bf((acc[r] + bias[c]) * gt);
        }
    }
}

// ---------------------------------------------------------------------------
// Kernel 3: fused flash attention + residual, 64-query block per workgroup.
// 8 waves; wave w owns S-tile (query subtile nt = w&3, key half mh = w>>2),
// i.e. every wave does softmax work, and PV channels [32w, 32w+32).
// Per key tile (32): 1 S-WMMA + 8 PV-WMMAs per wave.
// ---------------------------------------------------------------------------
__global__ __launch_bounds__(256) void attn_kernel(
    const unsigned short* __restrict__ qbuf,   // [B][N][32]
    const unsigned short* __restrict__ kbuf,   // [B][N][32]
    const unsigned short* __restrict__ vbuf,   // [B][C][N]
    const float* __restrict__ x,               // [B][C][N]
    const float* __restrict__ gamma,
    float* __restrict__ out)
{
    __shared__ unsigned short Pld[QBLK][MTILE];   // [query][key] bf16, 4 KB
    __shared__ float rmax[2][QBLK], rsum[2][QBLK];
    __shared__ float m_run[QBLK], l_run[QBLK], scl[QBLK], nmx[QBLK];

    const int b    = blockIdx.y;
    const int n0   = blockIdx.x * QBLK;
    const int lane = threadIdx.x & 31;
    const int w    = threadIdx.x >> 5;
    const int half = lane >> 4;
    const int l15  = lane & 15;
    const int nt   = w & 3;        // query subtile for S
    const int mh   = w >> 2;       // key half for S
    const int cb   = w * 32;       // PV channel base

    // Q A-fragment for this wave's S tile
    BFrag qf;
    {
        const unsigned short* qp =
            qbuf + ((size_t)b * NTOK + (n0 + nt * 16 + l15)) * C8_DIM;
        qf.lo = *(const u16x8*)(qp + 8 * half);
        qf.hi = *(const u16x8*)(qp + 16 + 8 * half);
    }
    if (threadIdx.x < QBLK) { m_run[threadIdx.x] = -INFINITY; l_run[threadIdx.x] = 0.0f; }
    __syncthreads();

    v8f o0[4] = {}, o1[4] = {};    // [query subtile] x {channels cb.., cb+16..}

    for (int m0 = 0; m0 < NTOK; m0 += MTILE) {
        // ---- S = Q^T K for this wave's 16x16 tile ----
        v8f s = {};
        {
            BFrag kf;
            const unsigned short* kp =
                kbuf + ((size_t)b * NTOK + (m0 + mh * 16 + l15)) * C8_DIM;
            kf.lo = *(const u16x8*)(kp + 8 * half);
            kf.hi = *(const u16x8*)(kp + 16 + 8 * half);
            s = wmma_bf16(qf, kf, s);
        }
        #pragma unroll
        for (int r = 0; r < 8; ++r) {
            float v = redmax16(s[r]);
            if (l15 == 0) rmax[mh][nt * 16 + r + 8 * half] = v;
        }
        __syncthreads();                       // B1: rmax ready

        if (threadIdx.x < QBLK) {
            int q = threadIdx.x;
            float mo = m_run[q];
            float mn = fmaxf(mo, fmaxf(rmax[0][q], rmax[1][q]));
            m_run[q] = mn;
            nmx[q]   = mn;
            scl[q]   = __expf(mo - mn);        // exp(-inf)=0 on first tile
        }
        __syncthreads();                       // B2: nmx/scl ready

        // ---- P = exp(S - m), row sums, stage P^T operand in LDS ----
        #pragma unroll
        for (int r = 0; r < 8; ++r) {
            int nq  = nt * 16 + r + 8 * half;
            float p = __expf(s[r] - nmx[nq]);
            Pld[nq][mh * 16 + l15] = f2bf(p);
            float v = redsum16(p);
            if (l15 == 0) rsum[mh][nq] = v;
        }
        // ---- V A-fragments for this wave's 32 channels (independent of P) ----
        BFrag vf0, vf1;
        const unsigned short* vp0 =
            vbuf + ((size_t)b * C_DIM + (cb + l15)) * NTOK + m0;
        const unsigned short* vp1 =
            vbuf + ((size_t)b * C_DIM + (cb + 16 + l15)) * NTOK + m0;
        vf0.lo = *(const u16x8*)(vp0 + 8 * half);
        vf0.hi = *(const u16x8*)(vp0 + 16 + 8 * half);
        vf1.lo = *(const u16x8*)(vp1 + 8 * half);
        vf1.hi = *(const u16x8*)(vp1 + 16 + 8 * half);
        if (m0 + MTILE < NTOK) {               // pull next key tile toward WGP
            __builtin_prefetch(vp0 + MTILE, 0, 3);
            __builtin_prefetch(vp1 + MTILE, 0, 3);
        }
        __syncthreads();                       // B3: P + rsum ready

        if (threadIdx.x < QBLK) {
            int q = threadIdx.x;
            l_run[q] = l_run[q] * scl[q] + rsum[0][q] + rsum[1][q];
        }
        // ---- O += V x P^T over all 4 query subtiles ----
        #pragma unroll
        for (int t = 0; t < 4; ++t) {
            float sc = scl[t * 16 + l15];
            #pragma unroll
            for (int r = 0; r < 8; ++r) { o0[t][r] *= sc; o1[t][r] *= sc; }
            BFrag pf;
            const unsigned short* pp = &Pld[t * 16 + l15][0];
            pf.lo = *(const u16x8*)(pp + 8 * half);
            pf.hi = *(const u16x8*)(pp + 16 + 8 * half);
            o0[t] = wmma_bf16(vf0, pf, o0[t]);
            o1[t] = wmma_bf16(vf1, pf, o1[t]);
        }
        __syncthreads();                       // protect Pld/stats for next tile
    }

    // ---- finalize: /l, gamma*O + x ----
    const float g  = gamma[0];
    const size_t bb = (size_t)b * C_DIM * NTOK;
    #pragma unroll
    for (int t = 0; t < 4; ++t) {
        const int   n    = n0 + t * 16 + l15;
        const float invl = 1.0f / l_run[t * 16 + l15];
        #pragma unroll
        for (int r = 0; r < 8; ++r) {
            int c0 = cb + r + 8 * half;
            size_t i0 = bb + (size_t)c0 * NTOK + n;
            out[i0] = g * (o0[t][r] * invl) + x[i0];
            int c1 = cb + 16 + r + 8 * half;
            size_t i1 = bb + (size_t)c1 * NTOK + n;
            out[i1] = g * (o1[t][r] * invl) + x[i1];
        }
    }
}

// ---------------------------------------------------------------------------
// Launch
// ---------------------------------------------------------------------------
extern "C" void kernel_launch(void* const* d_in, const int* in_sizes, int n_in,
                              void* d_out, int out_size, void* d_ws, size_t ws_size,
                              hipStream_t stream) {
    (void)in_sizes; (void)n_in; (void)out_size; (void)ws_size;
    const float* x     = (const float*)d_in[0];
    const float* gm    = (const float*)d_in[1];
    const float* Wq    = (const float*)d_in[2];
    const float* bq    = (const float*)d_in[3];
    const float* Wk    = (const float*)d_in[4];
    const float* bk    = (const float*)d_in[5];
    const float* Wv    = (const float*)d_in[6];
    const float* bv    = (const float*)d_in[7];
    const float* gamma = (const float*)d_in[8];
    float* out = (float*)d_out;

    char* ws = (char*)d_ws;
    float*          gate = (float*)ws;                                    // 64 KB
    unsigned short* qb = (unsigned short*)(ws + (64u << 10));             // 1 MB
    unsigned short* kb = (unsigned short*)(ws + (64u << 10) + (1u << 20));// 1 MB
    unsigned short* vb = (unsigned short*)(ws + (64u << 10) + (2u << 20));// 8 MB

    gate_kernel<<<64, 256, 0, stream>>>(gm, gate);
    qkv_kernel<<<dim3(NTOK / 16, 20, B_DIM), 32, 0, stream>>>(
        x, Wq, bq, Wk, bk, Wv, bv, gate, qb, kb, vb);
    attn_kernel<<<dim3(NTOK / QBLK, B_DIM), 256, 0, stream>>>(
        qb, kb, vb, x, gamma, out);
}
